// ForgettingController_55087250538933
// MI455X (gfx1250) — compile-verified, matrix operand order checked by reading, not run on hardware
//
#include <hip/hip_runtime.h>
#include <math.h>

#define Bn 256
#define Sn 65536
#define Dn 512
#define Hn 256
#define TOPK 3
#define CHUNK 256               // bank columns per block in sims kernel
#define NCHUNK (Sn / CHUNK)     // 256
#define ATROWS 64               // new_content rows per block in sims kernel
#define ASTR 520                // padded bf16 row stride (bank-conflict-free)
#define SSTR 257                // padded sims row stride (floats)
#define ESLOTS 128              // slots per block in erase kernel
#define NEBLK (Sn / ESLOTS)     // 512
#define CSTR 33                 // padded conf row stride
#define PROWS 16                // pair rows per block in pair-MLP kernel
#define PSTR 1026               // padded pair row stride (floats, even)
#define HSTR 257                // padded hidden row stride

typedef __attribute__((ext_vector_type(16))) __bf16 v16bf;
typedef __attribute__((ext_vector_type(8)))  float  v8f;
typedef __attribute__((ext_vector_type(2)))  float  v2f;

union FragB { v16bf v; __bf16 b[16]; unsigned u[8]; };
union AccF  { v8f v; float f[8]; };

__device__ __forceinline__ float sigm(float x) { return 1.0f / (1.0f + __expf(-x)); }

__device__ __forceinline__ float block_sum(float v, float* red, int t) {
  red[t] = v; __syncthreads();
  for (int s2 = 128; s2 > 0; s2 >>= 1) { if (t < s2) red[t] += red[t + s2]; __syncthreads(); }
  float r = red[0]; __syncthreads();
  return r;
}

// ---------------------------------------------------------------------------
// Kernel 1: sims tile via bf16 WMMA + per-chunk top-3
// grid = (NCHUNK, Bn/ATROWS), block = 256 (8 waves)
// ---------------------------------------------------------------------------
__global__ __launch_bounds__(256)
void sims_topk_kernel(const float* __restrict__ newc, const float* __restrict__ bank,
                      float* __restrict__ pval, int* __restrict__ pidx)
{
  // max(ATROWS*ASTR*2, ATROWS*SSTR*4) = max(66560, 65792)
  __shared__ __align__(16) char smem[ATROWS * ASTR * 2];
  unsigned short* At = (unsigned short*)smem;            // padded bf16 tile
  __bf16* Atb = (__bf16*)smem;
  float* sims = (float*)smem;                            // padded f32 (reused)

  const int tid   = threadIdx.x;
  const int sbase = blockIdx.x * CHUNK;
  const int rbase = blockIdx.y * ATROWS;

  for (int i = tid; i < ATROWS * Dn; i += 256) {
    const int r = i >> 9, c = i & (Dn - 1);
    Atb[r * ASTR + c] = (__bf16)newc[(size_t)(rbase + r) * Dn + c];
  }
  __syncthreads();

  const int wave = tid >> 5, lane = tid & 31;
  const int l15 = lane & 15;
  const int hi8 = (lane >> 4) * 8;          // lanes 16..31 take K offset +8

  const float* brow[2];
  brow[0] = bank + (size_t)(sbase + (wave * 2 + 0) * 16 + l15) * Dn;
  brow[1] = bank + (size_t)(sbase + (wave * 2 + 1) * 16 + l15) * Dn;

  AccF acc[2][4];
#pragma unroll
  for (int h = 0; h < 2; ++h)
#pragma unroll
    for (int bt = 0; bt < 4; ++bt)
#pragma unroll
      for (int r = 0; r < 8; ++r) acc[h][bt].f[r] = 0.0f;

  for (int ks = 0; ks < 16; ++ks) {
    const int k0 = ks * 32 + hi8;
    __builtin_prefetch(brow[0] + ks * 32 + 64, 0, 1);
    __builtin_prefetch(brow[1] + ks * 32 + 64, 0, 1);

    FragB af[4];
#pragma unroll
    for (int bt = 0; bt < 4; ++bt) {
      const unsigned* ar = (const unsigned*)(At + (size_t)(bt * 16 + l15) * ASTR + k0);
      af[bt].u[0] = ar[0]; af[bt].u[1] = ar[1]; af[bt].u[2] = ar[2]; af[bt].u[3] = ar[3];
      af[bt].u[4] = ar[8]; af[bt].u[5] = ar[9]; af[bt].u[6] = ar[10]; af[bt].u[7] = ar[11];
    }
#pragma unroll
    for (int h = 0; h < 2; ++h) {
      FragB bfr;
#pragma unroll
      for (int c = 0; c < 2; ++c) {
        const float* p = brow[h] + k0 + c * 16;
#pragma unroll
        for (int j = 0; j < 8; ++j)
          bfr.b[c * 8 + j] = (__bf16)p[j];               // v_cvt_pk_bf16_f32
      }
#pragma unroll
      for (int bt = 0; bt < 4; ++bt)
        acc[h][bt].v = __builtin_amdgcn_wmma_f32_16x16x32_bf16(
            false, af[bt].v, false, bfr.v, (short)0, acc[h][bt].v, false, false);
    }
  }
  __syncthreads();   // A tile dead; reuse LDS as sims buffer

#pragma unroll
  for (int h = 0; h < 2; ++h) {
    const int col = (wave * 2 + h) * 16 + l15;
#pragma unroll
    for (int bt = 0; bt < 4; ++bt)
#pragma unroll
      for (int r = 0; r < 8; ++r)
        sims[(size_t)(bt * 16 + r + hi8) * SSTR + col] = acc[h][bt].f[r];
  }
  __syncthreads();

  if (tid < ATROWS) {
    float v0 = -1e30f, v1 = -1e30f, v2 = -1e30f; int i0 = 0, i1 = 0, i2 = 0;
    const float* row = sims + (size_t)tid * SSTR;
    for (int c = 0; c < CHUNK; ++c) {
      const float v = row[c]; const int gi = sbase + c;
      if (v > v0)      { v2 = v1; i2 = i1; v1 = v0; i1 = i0; v0 = v; i0 = gi; }
      else if (v > v1) { v2 = v1; i2 = i1; v1 = v; i1 = gi; }
      else if (v > v2) { v2 = v; i2 = gi; }
    }
    const size_t o = ((size_t)(rbase + tid) * NCHUNK + blockIdx.x) * 3;
    pval[o] = v0; pval[o + 1] = v1; pval[o + 2] = v2;
    pidx[o] = i0; pidx[o + 1] = i1; pidx[o + 2] = i2;
  }
}

// ---------------------------------------------------------------------------
// Kernel 2: merge chunk partials -> global top-3 per row.  <<<1,256>>>
// ---------------------------------------------------------------------------
__global__ __launch_bounds__(256)
void topk_merge_kernel(const float* __restrict__ pval, const int* __restrict__ pidx,
                       float* __restrict__ tv, int* __restrict__ ti, float* __restrict__ maxsim)
{
  const int b = threadIdx.x;
  float v0 = -1e30f, v1 = -1e30f, v2 = -1e30f; int i0 = 0, i1 = 0, i2 = 0;
  const size_t base = (size_t)b * NCHUNK * 3;
  for (int j = 0; j < NCHUNK * 3; ++j) {
    const float v = pval[base + j]; const int gi = pidx[base + j];
    if (v > v0)      { v2 = v1; i2 = i1; v1 = v0; i1 = i0; v0 = v; i0 = gi; }
    else if (v > v1) { v2 = v1; i2 = i1; v1 = v; i1 = gi; }
    else if (v > v2) { v2 = v; i2 = gi; }
  }
  tv[b * 3] = v0; tv[b * 3 + 1] = v1; tv[b * 3 + 2] = v2;
  ti[b * 3] = i0; ti[b * 3 + 1] = i1; ti[b * 3 + 2] = i2;
  maxsim[b] = v0;
}

// ---------------------------------------------------------------------------
// Kernel 3: store-score MLP (one block per batch row)
// ---------------------------------------------------------------------------
__global__ __launch_bounds__(256)
void store_score_kernel(const float* __restrict__ newc, const float* __restrict__ query,
                        const float* __restrict__ w1, const float* __restrict__ b1,
                        const float* __restrict__ lng, const float* __restrict__ lnb,
                        const float* __restrict__ w2, const float* __restrict__ b2,
                        const float* __restrict__ wn, const float* __restrict__ bnv,
                        const float* __restrict__ wg, const float* __restrict__ bg,
                        float* __restrict__ score)
{
  __shared__ float comb[2 * Dn];
  __shared__ float h[Hn];
  __shared__ float red[256];
  const int b = blockIdx.x, t = threadIdx.x;
  for (int i = t; i < Dn; i += 256) {
    comb[i]      = newc [(size_t)b * Dn + i];
    comb[Dn + i] = query[(size_t)b * Dn + i];
  }
  __syncthreads();
  float a = b1[t];
  for (int k = 0; k < 2 * Dn; ++k) a += comb[k] * w1[(size_t)k * Hn + t];
  const float mean = block_sum(a, red, t) / (float)Hn;
  const float d0 = a - mean;
  const float var = block_sum(d0 * d0, red, t) / (float)Hn;
  h[t] = d0 * __frsqrt_rn(var + 1e-5f) * lng[t] + lnb[t];
  __syncthreads();
  float fv = 0.0f;
  if (t < Hn / 2) {
    float a2 = b2[t];
    for (int k = 0; k < Hn; ++k) a2 += fmaxf(h[k], 0.0f) * w2[(size_t)k * (Hn / 2) + t];
    float an = bnv[t];
    for (int k = 0; k < Dn; ++k) an += comb[k] * wn[(size_t)k * (Hn / 2) + t];
    fv = (fmaxf(a2, 0.0f) + sigm(an)) * wg[t];
  }
  const float gs = block_sum(fv, red, t);
  if (t == 0) score[b] = sigm(gs + bg[0]);
}

// ---------------------------------------------------------------------------
// Kernel 4: erase gate via fp32 WMMA (V_WMMA_F32_16X16X4_F32).
// grid = Sn/ESLOTS, block = 256 (8 waves x 16 slots = 128 slots/block).
// ---------------------------------------------------------------------------
__global__ __launch_bounds__(256)
void erase_kernel(const float* __restrict__ bank, const float* __restrict__ at,
                  const float* __restrict__ wl, const float* __restrict__ bl,
                  const float* __restrict__ wc, const float* __restrict__ bc,
                  const float* __restrict__ we, const float* __restrict__ be,
                  const float* __restrict__ decay, const int* __restrict__ stepp,
                  float* __restrict__ erase_out,
                  int* __restrict__ winner, float* __restrict__ sw, int* __restrict__ bw,
                  float* __restrict__ eargv, int* __restrict__ eargi)
{
  __shared__ float wB[Dn * 32];               // 64 KB: w_conf staged
  __shared__ float conf[ESLOTS * CSTR];       // padded (conflict-free phase 2)
  __shared__ float rv[ESLOTS];
  __shared__ int   ri[ESLOTS];

  const int tid = threadIdx.x;
  for (int i = tid; i < Dn * 32; i += 256) wB[i] = wc[i];
  __syncthreads();

  const int wave = tid >> 5, lane = tid & 31;
  const int l15 = lane & 15;
  const int hi2 = (lane >> 4) * 2;            // lanes 16..31: K offset +2
  const int hi8 = (lane >> 4) * 8;
  const int srow = blockIdx.x * ESLOTS + wave * 16 + l15;
  const float* arow = bank + (size_t)srow * Dn + hi2;

  AccF acc0, acc1;
#pragma unroll
  for (int r = 0; r < 8; ++r) { acc0.f[r] = 0.0f; acc1.f[r] = 0.0f; }

  for (int k0 = 0; k0 < Dn; k0 += 4) {
    __builtin_prefetch(arow + k0 + 32, 0, 1);
    const v2f afr = { arow[k0], arow[k0 + 1] };            // K=k0(+hi2), k0+1(+hi2)
    const int kb = k0 + hi2;
    const v2f bf0 = { wB[kb * 32 + l15],      wB[(kb + 1) * 32 + l15] };
    const v2f bf1 = { wB[kb * 32 + 16 + l15], wB[(kb + 1) * 32 + 16 + l15] };
    acc0.v = __builtin_amdgcn_wmma_f32_16x16x4_f32(
        false, afr, false, bf0, (short)0, acc0.v, false, false);
    acc1.v = __builtin_amdgcn_wmma_f32_16x16x4_f32(
        false, afr, false, bf1, (short)0, acc1.v, false, false);
  }
#pragma unroll
  for (int r = 0; r < 8; ++r) {
    conf[(wave * 16 + r + hi8) * CSTR + l15]      = acc0.f[r];
    conf[(wave * 16 + r + hi8) * CSTR + 16 + l15] = acc1.f[r];
  }
  __syncthreads();

  if (tid < ESLOTS) {
    const int s = blockIdx.x * ESLOTS + tid;
    const float t = ((float)(*stepp) - at[s]) / 1000.0f;
    float e = be[0];
    for (int j = 0; j < 32; ++j) {
      const float lr = fmaxf(t * wl[j] + bl[j], 0.0f);
      e += lr * we[j] + sigm(conf[tid * CSTR + j] + bc[j]) * we[32 + j];
    }
    const float ev = sigm(e) * (1.0f - __expf(-decay[0] * t));
    erase_out[s] = ev;
    winner[s] = 0; sw[s] = 0.0f; bw[s] = 0;
    rv[tid] = ev; ri[tid] = s;
  }
  __syncthreads();
  for (int s2 = ESLOTS / 2; s2 > 0; s2 >>= 1) {
    if (tid < s2 && (rv[tid + s2] > rv[tid])) { rv[tid] = rv[tid + s2]; ri[tid] = ri[tid + s2]; }
    __syncthreads();
  }
  if (tid == 0) { eargv[blockIdx.x] = rv[0]; eargi[blockIdx.x] = ri[0]; }
}

// ---------------------------------------------------------------------------
// Kernel 5: dd/ds conflict MLPs via fp32 WMMA.
// grid = (Bn*TOPK)/PROWS = 48, block = 256 (8 waves x 2 col-tiles).
// GEMM1 (1024->256, bias folded into accumulator) on the matrix pipe,
// relu'd H to LDS, scalar sigmoid GEMV head.
// ---------------------------------------------------------------------------
__global__ __launch_bounds__(256)
void pair_mlp_kernel(const float* __restrict__ newc, const float* __restrict__ bank,
                     const int* __restrict__ ti,
                     const float* __restrict__ wd1, const float* __restrict__ bd1,
                     const float* __restrict__ wd2, const float* __restrict__ bd2,
                     const float* __restrict__ ws1, const float* __restrict__ bs1,
                     const float* __restrict__ ws2, const float* __restrict__ bs2,
                     float* __restrict__ dd, float* __restrict__ dsv)
{
  __shared__ float P[PROWS * PSTR];           // padded pair rows (~66 KB)
  __shared__ float Hs[PROWS * HSTR];          // padded hidden (~16 KB)
  __shared__ int idxs[PROWS];
  const int tid = threadIdx.x;
  const int p0 = blockIdx.x * PROWS;
  if (tid < PROWS) idxs[tid] = ti[p0 + tid];
  __syncthreads();
  for (int i = tid; i < PROWS * 2 * Dn; i += 256) {
    const int r = i >> 10, c = i & (2 * Dn - 1);
    P[r * PSTR + c] = (c < Dn) ? newc[(size_t)((p0 + r) / TOPK) * Dn + c]
                               : bank[(size_t)idxs[r] * Dn + (c - Dn)];
  }
  __syncthreads();

  const int wave = tid >> 5, lane = tid & 31;
  const int l15 = lane & 15;
  const int hi2 = (lane >> 4) * 2;
  const int hi8 = (lane >> 4) * 8;

  for (int net = 0; net < 2; ++net) {
    const float* w1 = net ? ws1 : wd1;
    const float* b1 = net ? bs1 : bd1;
#pragma unroll
    for (int h = 0; h < 2; ++h) {
      const int col = (wave * 2 + h) * 16 + l15;
      AccF acc;
      const float bias = b1[col];
#pragma unroll
      for (int r = 0; r < 8; ++r) acc.f[r] = bias;
      for (int k0 = 0; k0 < 2 * Dn; k0 += 4) {
        const int kb = k0 + hi2;
        const v2f afr = { P[l15 * PSTR + kb], P[l15 * PSTR + kb + 1] };
        const v2f bfr = { w1[(size_t)kb * Hn + col], w1[(size_t)(kb + 1) * Hn + col] };
        acc.v = __builtin_amdgcn_wmma_f32_16x16x4_f32(
            false, afr, false, bfr, (short)0, acc.v, false, false);
      }
#pragma unroll
      for (int r = 0; r < 8; ++r)
        Hs[(r + hi8) * HSTR + col] = fmaxf(acc.f[r], 0.0f);
    }
    __syncthreads();
    if (tid < PROWS) {
      const float* w2 = net ? ws2 : wd2;
      float g = net ? bs2[0] : bd2[0];
      for (int c = 0; c < Hn; ++c) g += Hs[tid * HSTR + c] * w2[c];
      (net ? dsv : dd)[p0 + tid] = sigm(g);
    }
    __syncthreads();
  }
}

// ---------------------------------------------------------------------------
// Kernel 6: single-block resolve: erase argmax, means, scatter-max conflicts,
// drifted_new row 0, scalar outputs.
// ---------------------------------------------------------------------------
__global__ __launch_bounds__(256)
void resolve_kernel(const float* __restrict__ newc, const float* __restrict__ bank,
                    const float* __restrict__ tv, const int* __restrict__ ti,
                    const float* __restrict__ dd, const float* __restrict__ dsv,
                    const float* __restrict__ score, const float* __restrict__ maxsim,
                    const float* __restrict__ eargv, const int* __restrict__ eargi,
                    int* __restrict__ winner, float* __restrict__ sw, int* __restrict__ bw,
                    float* __restrict__ dn0, int* __restrict__ scal, float* __restrict__ dout)
{
  __shared__ float rv[256];
  __shared__ int   ri[256];
  __shared__ int cnt_sh, eidx_sh, cond_sh, any0_sh, old0_sh;
  __shared__ float s0_sh;
  const int t = threadIdx.x;
  if (t == 0) cnt_sh = 0;

  // erase argmax (first max wins)
  float best = -1e30f; int bidx = 0x7fffffff;
  for (int j = t; j < NEBLK; j += 256) {
    const float v = eargv[j]; const int ii = eargi[j];
    if (v > best || (v == best && ii < bidx)) { best = v; bidx = ii; }
  }
  rv[t] = best; ri[t] = bidx; __syncthreads();
  for (int s2 = 128; s2 > 0; s2 >>= 1) {
    if (t < s2 && (rv[t + s2] > rv[t] || (rv[t + s2] == rv[t] && ri[t + s2] < ri[t]))) {
      rv[t] = rv[t + s2]; ri[t] = ri[t + s2];
    }
    __syncthreads();
  }
  if (t == 0) eidx_sh = ri[0];
  __syncthreads();

  // means
  const float smean = block_sum(score[t], rv, t) / 256.0f;
  const float ms = maxsim[t];
  const float novm = block_sum(1.0f - sigm(ms * 5.0f), rv, t) / 256.0f;
  const float relm = block_sum(sigm(ms), rv, t) / 256.0f;
  if (t == 0) {
    dout[0] = smean; dout[1] = novm; dout[2] = relm;
    cond_sh = (smean > 0.5f) ? 1 : 0;
  }
  __syncthreads();

  // conflict scatter-max (priority = flat index + 1 -> torch last-write-wins)
  for (int i = t; i < Bn * TOPK; i += 256) {
    const float v = tv[i];
    if (v > 0.7f && v < 0.99f && dd[i] > 0.5f) atomicMax(&winner[ti[i]], i + 1);
  }
  __threadfence(); __syncthreads();
  for (int i = t; i < Bn * TOPK; i += 256) {
    const float v = tv[i];
    if (v > 0.7f && v < 0.99f && dd[i] > 0.5f) {
      const int slot = ti[i];
      if (atomicMax(&winner[slot], 0) == i + 1) {
        sw[slot] = dsv[i]; bw[slot] = i / TOPK; atomicAdd(&cnt_sh, 1);
      }
    }
  }
  __syncthreads();

  if (t == 0) {
    int lastk = -1;
    for (int k = 0; k < TOPK; ++k) {
      const float v = tv[k];
      if (v > 0.7f && v < 0.99f && dd[k] > 0.5f) lastk = k;
    }
    any0_sh = (lastk >= 0) ? 1 : 0;
    s0_sh   = (lastk >= 0) ? dsv[lastk] : 0.0f;
    old0_sh = (lastk >= 0) ? ti[lastk] : 0;
    dout[3 + Sn] = (float)cnt_sh;                 // full_conflict.sum()
    scal[0] = eidx_sh; scal[1] = cond_sh; scal[2] = cnt_sh;
  }
  __syncthreads();
  for (int d2 = t; d2 < Dn; d2 += 256) {
    const float n = newc[d2];
    float val = n;
    if (any0_sh) {
      const float o = bank[(size_t)old0_sh * Dn + d2];
      val = n + s0_sh * (o - n) * 0.5f;           // (1-s)*n + s*(n+o)/2
    }
    dn0[d2] = val;
  }
}

// ---------------------------------------------------------------------------
// Kernel 7: build updated_memory / updated_access + |delta| partials. grid=Sn
// ---------------------------------------------------------------------------
__global__ __launch_bounds__(256)
void update_kernel(const float* __restrict__ bank, const float* __restrict__ newc,
                   const float* __restrict__ at, const int* __restrict__ stepp,
                   const int* __restrict__ winner, const float* __restrict__ sw,
                   const int* __restrict__ bw, const float* __restrict__ dn0,
                   const int* __restrict__ scal,
                   float* __restrict__ outmem, float* __restrict__ outacc,
                   float* __restrict__ recpart)
{
  __shared__ float red[256];
  const int s = blockIdx.x, t = threadIdx.x;
  const int has = winner[s] > 0;
  const float swv = has ? sw[s] : 0.0f;
  const int bwi = has ? bw[s] : 0;
  const bool iser = (s == scal[0]) && (scal[1] != 0);
  float psum = 0.0f;
#pragma unroll
  for (int rr = 0; rr < 2; ++rr) {
    const int d2 = t + rr * 256;
    const float o = bank[(size_t)s * Dn + d2];
    float val = o;
    if (has)  val = o + swv * (newc[(size_t)bwi * Dn + d2] - o) * 0.5f;
    if (iser) val = dn0[d2];
    outmem[(size_t)s * Dn + d2] = val;
    psum += fabsf(val - o);
  }
  const float bs = block_sum(psum, red, t);
  if (t == 0) {
    recpart[s] = bs;
    outacc[s] = iser ? (float)(*stepp) : at[s];
  }
}

// ---------------------------------------------------------------------------
// Kernel 8: new_decay.  <<<1,256>>>
// ---------------------------------------------------------------------------
__global__ __launch_bounds__(256)
void finalize_kernel(const float* __restrict__ recpart, const float* __restrict__ decay,
                     float* __restrict__ dout)
{
  __shared__ float red[256];
  const int t = threadIdx.x;
  float s = 0.0f;
  for (int j = t; j < Sn; j += 256) s += recpart[j];
  const float tot = block_sum(s, red, t);
  if (t == 0) {
    const float recent = tot / ((float)Sn * (float)Dn);
    const float adj = (0.1f * recent > 0.1f) ? -0.001f : 0.001f;
    dout[4 + Sn] = fminf(fmaxf(decay[0] + adj, 0.001f), 0.1f);
  }
}

// ---------------------------------------------------------------------------
extern "C" void kernel_launch(void* const* d_in, const int* in_sizes, int n_in,
                              void* d_out, int out_size, void* d_ws, size_t ws_size,
                              hipStream_t stream)
{
  (void)in_sizes; (void)n_in; (void)out_size; (void)ws_size;
  const float* newc   = (const float*)d_in[0];
  const float* query  = (const float*)d_in[1];
  const float* bank   = (const float*)d_in[2];
  const float* at     = (const float*)d_in[3];
  const float* w_rel1 = (const float*)d_in[4];
  const float* b_rel1 = (const float*)d_in[5];
  const float* ln_g   = (const float*)d_in[6];
  const float* ln_b   = (const float*)d_in[7];
  const float* w_rel2 = (const float*)d_in[8];
  const float* b_rel2 = (const float*)d_in[9];
  const float* w_nov  = (const float*)d_in[10];
  const float* b_nov  = (const float*)d_in[11];
  const float* w_gate = (const float*)d_in[12];
  const float* b_gate = (const float*)d_in[13];
  const float* w_lru  = (const float*)d_in[14];
  const float* b_lru  = (const float*)d_in[15];
  const float* w_conf = (const float*)d_in[16];
  const float* b_conf = (const float*)d_in[17];
  const float* w_er   = (const float*)d_in[18];
  const float* b_er   = (const float*)d_in[19];
  const float* w_dd1  = (const float*)d_in[20];
  const float* b_dd1  = (const float*)d_in[21];
  const float* w_dd2  = (const float*)d_in[22];
  const float* b_dd2  = (const float*)d_in[23];
  const float* w_ds1  = (const float*)d_in[24];
  const float* b_ds1  = (const float*)d_in[25];
  const float* w_ds2  = (const float*)d_in[26];
  const float* b_ds2  = (const float*)d_in[27];
  const float* decay  = (const float*)d_in[28];
  const int*   stepp  = (const int*)d_in[29];

  char* w = (char*)d_ws;
  size_t o = 0;
  auto alloc = [&](size_t bytes) -> char* {
    char* p = w + o; o += (bytes + 255) & ~(size_t)255; return p;
  };
  float* pval   = (float*)alloc(sizeof(float) * Bn * NCHUNK * 3);
  int*   pidx   = (int*)  alloc(sizeof(int)   * Bn * NCHUNK * 3);
  float* tv     = (float*)alloc(sizeof(float) * Bn * TOPK);
  int*   ti     = (int*)  alloc(sizeof(int)   * Bn * TOPK);
  float* maxsim = (float*)alloc(sizeof(float) * Bn);
  float* score  = (float*)alloc(sizeof(float) * Bn);
  float* ddv    = (float*)alloc(sizeof(float) * Bn * TOPK);
  float* dsv    = (float*)alloc(sizeof(float) * Bn * TOPK);
  int*   winner = (int*)  alloc(sizeof(int)   * Sn);
  float* sww    = (float*)alloc(sizeof(float) * Sn);
  int*   bww    = (int*)  alloc(sizeof(int)   * Sn);
  float* eargv  = (float*)alloc(sizeof(float) * NEBLK);
  int*   eargi  = (int*)  alloc(sizeof(int)   * NEBLK);
  float* dn0    = (float*)alloc(sizeof(float) * Dn);
  float* recp   = (float*)alloc(sizeof(float) * Sn);
  int*   scal   = (int*)  alloc(sizeof(int)   * 16);

  float* dout      = (float*)d_out;
  float* out_erase = dout + 3;
  float* out_mem   = dout + 5 + (size_t)Sn;
  float* out_acc   = out_mem + (size_t)Sn * Dn;

  sims_topk_kernel<<<dim3(NCHUNK, Bn / ATROWS), 256, 0, stream>>>(newc, bank, pval, pidx);
  topk_merge_kernel<<<1, 256, 0, stream>>>(pval, pidx, tv, ti, maxsim);
  store_score_kernel<<<Bn, 256, 0, stream>>>(newc, query, w_rel1, b_rel1, ln_g, ln_b,
                                             w_rel2, b_rel2, w_nov, b_nov, w_gate, b_gate, score);
  erase_kernel<<<Sn / ESLOTS, 256, 0, stream>>>(bank, at, w_lru, b_lru, w_conf, b_conf,
                                                w_er, b_er, decay, stepp, out_erase,
                                                winner, sww, bww, eargv, eargi);
  pair_mlp_kernel<<<(Bn * TOPK) / PROWS, 256, 0, stream>>>(newc, bank, ti,
                                                 w_dd1, b_dd1, w_dd2, b_dd2,
                                                 w_ds1, b_ds1, w_ds2, b_ds2, ddv, dsv);
  resolve_kernel<<<1, 256, 0, stream>>>(newc, bank, tv, ti, ddv, dsv, score, maxsim,
                                        eargv, eargi, winner, sww, bww, dn0, scal, dout);
  update_kernel<<<Sn, 256, 0, stream>>>(bank, newc, at, stepp, winner, sww, bww, dn0, scal,
                                        out_mem, out_acc, recp);
  finalize_kernel<<<1, 256, 0, stream>>>(recp, decay, dout);
}
